// AttentionLayer_77429670412962
// MI455X (gfx1250) — compile-verified, matrix operand order checked by reading, not run on hardware
//
#include <hip/hip_runtime.h>
#include <hip/hip_bf16.h>

// Problem sizes (fixed by the reference)
#define SB 64      // batch
#define SS 512     // sequence / score dim
#define SF 256     // feature dim
#define MT 32      // M rows per block (2 x 16-row WMMA tiles)

typedef __attribute__((ext_vector_type(16))) __bf16 v16bf;
typedef __attribute__((ext_vector_type(8)))  float  v8f;

union Frag16 {          // 16 bf16 = 32 bytes = two 16-byte halves
    v16bf v;
    uint4 u[2];
};

// Load one 16x32 bf16 fragment for this lane from a row-major matrix.
__device__ __forceinline__ v16bf load_frag(const __bf16* __restrict__ base,
                                           int row, int rowStride,
                                           int kb1, int kb2) {
    Frag16 f;
    f.u[0] = *(const uint4*)(base + row * rowStride + kb1);
    f.u[1] = *(const uint4*)(base + row * rowStride + kb2);
    return f.v;
}

// ---------------------------------------------------------------------------
// Prep: dst[n*K + k] = (bf16) src[k*N + n]   (W^T, bf16, N-major)
// ---------------------------------------------------------------------------
__global__ void transpose_to_bf16(const float* __restrict__ src,
                                  __bf16* __restrict__ dst,
                                  int K, int N) {
    int idx = blockIdx.x * blockDim.x + threadIdx.x;
    if (idx >= K * N) return;
    int n = idx / K;
    int k = idx - n * K;
    dst[idx] = (__bf16)src[k * N + n];
}

// ---------------------------------------------------------------------------
// Fused: scores = x1@Wq + x2@Wk ; e = exp(tanh(scores)) ;
//        out = diag(1/rowsum(e)) * (e @ Wv) * x1
// One block = 8 wave32 = one 32-row tile, all 512 columns.
// Wave w owns columns [64w, 64w+64) (4 N-tiles) x 2 M-tiles.
// Each B fragment feeds two WMMAs (both M-tiles).
// ---------------------------------------------------------------------------
__launch_bounds__(256)
__global__ void attn_fused(const float* __restrict__ x1,
                           const float* __restrict__ x2,
                           const __bf16* __restrict__ WqT,   // [512][512]
                           const __bf16* __restrict__ WkT,   // [512][256]
                           const __bf16* __restrict__ WvT,   // [512][512]
                           float* __restrict__ out) {
    __shared__ __bf16 bufA[MT * SS];     // 32 KB: x1 tile (phase 1) / e tile (phase 2)
    __shared__ __bf16 bufB[MT * SF];     // 16 KB: x2 tile
    __shared__ float  rowsum[MT];        // softmax denominators

    const int tid  = threadIdx.x;
    const int wave = tid >> 5;
    const int lane = tid & 31;

    const int blk    = blockIdx.x;        // 0 .. 1023
    const int b      = blk >> 4;          // batch
    const int s_base = (blk & 15) << 5;   // row tile base (multiples of 32)

    const float* x1_tile = x1 + ((size_t)b * SS + s_base) * SS;  // 32*512 contiguous f32
    const float* x2_tile = x2 + ((size_t)b * SS + s_base) * SF;  // 32*256 contiguous f32

    // ---- stage x1 tile: 4096 float4 -> bf16 LDS (16 per thread) ----
    #pragma unroll
    for (int i = 0; i < 16; ++i) {
        int j = tid + i * 256;
        float4 f = ((const float4*)x1_tile)[j];
        union { __bf16 h[4]; uint2 u; } pk;
        pk.h[0] = (__bf16)f.x; pk.h[1] = (__bf16)f.y;
        pk.h[2] = (__bf16)f.z; pk.h[3] = (__bf16)f.w;
        *(uint2*)&bufA[j * 4] = pk.u;
    }
    // ---- stage x2 tile: 2048 float4 (8 per thread) ----
    #pragma unroll
    for (int i = 0; i < 8; ++i) {
        int j = tid + i * 256;
        float4 f = ((const float4*)x2_tile)[j];
        union { __bf16 h[4]; uint2 u; } pk;
        pk.h[0] = (__bf16)f.x; pk.h[1] = (__bf16)f.y;
        pk.h[2] = (__bf16)f.z; pk.h[3] = (__bf16)f.w;
        *(uint2*)&bufB[j * 4] = pk.u;
    }
    if (tid < MT) rowsum[tid] = 0.0f;

    const int colb  = wave << 6;                 // wave's column base
    const int rowA  = lane & 15;                 // A-row within a 16-row M-tile
    const int kh1   = (lane < 16) ? 0 : 8;       // K-half bases per ISA bf16 A layout
    const int kh2   = kh1 + 16;
    const int mofs  = (lane < 16) ? 0 : 8;       // C/D row offset per ISA layout

    // warm the Wv^T columns this wave will need later (global_prefetch_b8)
    __builtin_prefetch(&WvT[(size_t)(colb + lane) * SS], 0, 0);
    __builtin_prefetch(&WvT[(size_t)(colb + 32 + lane) * SS], 0, 0);

    __syncthreads();

    // ---- GEMM1: scores tile (2 M-tiles x 4 N-tiles), f32 accumulate ----
    v8f acc[2][4] = {};
    for (int kk = 0; kk < 16; ++kk) {            // x1 @ Wq, K = 512
        int k0 = kk << 5;
        v16bf a0 = load_frag(bufA,           rowA, SS, k0 + kh1, k0 + kh2);
        v16bf a1 = load_frag(bufA + 16 * SS, rowA, SS, k0 + kh1, k0 + kh2);
        #pragma unroll
        for (int t = 0; t < 4; ++t) {
            int n = colb + (t << 4) + (lane & 15);
            v16bf bf = load_frag(WqT, n, SS, k0 + kh1, k0 + kh2);
            acc[0][t] = __builtin_amdgcn_wmma_f32_16x16x32_bf16(
                false, a0, false, bf, (short)0, acc[0][t], false, false);
            acc[1][t] = __builtin_amdgcn_wmma_f32_16x16x32_bf16(
                false, a1, false, bf, (short)0, acc[1][t], false, false);
        }
    }
    for (int kk = 0; kk < 8; ++kk) {             // + x2 @ Wk, K = 256
        int k0 = kk << 5;
        v16bf a0 = load_frag(bufB,           rowA, SF, k0 + kh1, k0 + kh2);
        v16bf a1 = load_frag(bufB + 16 * SF, rowA, SF, k0 + kh1, k0 + kh2);
        #pragma unroll
        for (int t = 0; t < 4; ++t) {
            int n = colb + (t << 4) + (lane & 15);
            v16bf bf = load_frag(WkT, n, SF, k0 + kh1, k0 + kh2);
            acc[0][t] = __builtin_amdgcn_wmma_f32_16x16x32_bf16(
                false, a0, false, bf, (short)0, acc[0][t], false, false);
            acc[1][t] = __builtin_amdgcn_wmma_f32_16x16x32_bf16(
                false, a1, false, bf, (short)0, acc[1][t], false, false);
        }
    }

    // all waves done reading bufA/bufB before we overwrite bufA with e
    __syncthreads();

    // ---- e = exp(tanh(score)); store UNNORMALIZED to LDS, accumulate row sums.
    // tanh bounds scores to [-1,1] -> exp in [1/e, e]; no max subtraction needed.
    // softmax division is linear: applied as diag(1/rowsum) in the epilogue.
    float psum[2][8] = {};
    #pragma unroll
    for (int mt = 0; mt < 2; ++mt)
        #pragma unroll
        for (int t = 0; t < 4; ++t) {
            int col = colb + (t << 4) + (lane & 15);
            #pragma unroll
            for (int r = 0; r < 8; ++r) {
                float e = __expf(tanhf(acc[mt][t][r]));
                bufA[((mt << 4) + r + mofs) * SS + col] = (__bf16)e;
                psum[mt][r] += e;
            }
        }
    #pragma unroll
    for (int mt = 0; mt < 2; ++mt)
        #pragma unroll
        for (int r = 0; r < 8; ++r)
            atomicAdd(&rowsum[(mt << 4) + r + mofs], psum[mt][r]);   // ds_add_f32
    __syncthreads();

    // ---- GEMM3: e @ Wv, K = 512 ----
    v8f acc2[2][4] = {};
    for (int kk = 0; kk < 16; ++kk) {
        int k0 = kk << 5;
        v16bf a0 = load_frag(bufA,           rowA, SS, k0 + kh1, k0 + kh2);
        v16bf a1 = load_frag(bufA + 16 * SS, rowA, SS, k0 + kh1, k0 + kh2);
        #pragma unroll
        for (int t = 0; t < 4; ++t) {
            int n = colb + (t << 4) + (lane & 15);
            v16bf bf = load_frag(WvT, n, SS, k0 + kh1, k0 + kh2);
            acc2[0][t] = __builtin_amdgcn_wmma_f32_16x16x32_bf16(
                false, a0, false, bf, (short)0, acc2[0][t], false, false);
            acc2[1][t] = __builtin_amdgcn_wmma_f32_16x16x32_bf16(
                false, a1, false, bf, (short)0, acc2[1][t], false, false);
        }
    }

    // ---- epilogue: softmax scale * x1, write out (L2-norm applied in-place later)
    float inv[2][8];
    #pragma unroll
    for (int mt = 0; mt < 2; ++mt)
        #pragma unroll
        for (int r = 0; r < 8; ++r)
            inv[mt][r] = 1.0f / rowsum[(mt << 4) + r + mofs];

    #pragma unroll
    for (int mt = 0; mt < 2; ++mt)
        #pragma unroll
        for (int t = 0; t < 4; ++t) {
            int col = colb + (t << 4) + (lane & 15);
            #pragma unroll
            for (int r = 0; r < 8; ++r) {
                int m = (mt << 4) + r + mofs;
                size_t idx = ((size_t)b * SS + s_base + m) * SS + col;
                out[idx] = acc2[mt][t][r] * inv[mt][r] * x1[idx];
            }
        }
}

// ---------------------------------------------------------------------------
// In-place L2 normalization over axis=1 (the s dimension), per (b, t).
// Grid (64, 2): block (b, h) owns columns [256h, 256h+256), 1 per thread.
// Each 1 MB batch slice is L2-resident across the two passes.
// ---------------------------------------------------------------------------
__global__ void l2norm_inplace(float* __restrict__ out) {
    const int b = blockIdx.x;
    float* p = out + (size_t)b * SS * SS;
    const int c = blockIdx.y * 256 + threadIdx.x;

    float a0 = 0.0f;
    for (int s = 0; s < SS; ++s) {
        float v = p[s * SS + c];
        a0 += v * v;
    }
    float r0 = rsqrtf(fmaxf(a0, 1e-12f));
    for (int s = 0; s < SS; ++s)
        p[s * SS + c] *= r0;
}

// ---------------------------------------------------------------------------
extern "C" void kernel_launch(void* const* d_in, const int* in_sizes, int n_in,
                              void* d_out, int out_size, void* d_ws, size_t ws_size,
                              hipStream_t stream) {
    const float* x1 = (const float*)d_in[0];   // [64,512,512]
    const float* x2 = (const float*)d_in[1];   // [64,512,256]
    const float* Wq = (const float*)d_in[2];   // [512,512]
    const float* Wk = (const float*)d_in[3];   // [256,512]
    const float* Wv = (const float*)d_in[4];   // [512,512]
    float* out = (float*)d_out;                // [64,512,512]

    // workspace: bf16 transposed weights only (~1.25 MB, L2-resident)
    char* ws = (char*)d_ws;
    __bf16* WqT = (__bf16*)(ws);                          // 512*512*2 = 512 KB
    __bf16* WkT = (__bf16*)(ws + 512 * 1024);             // 512*256*2 = 256 KB
    __bf16* WvT = (__bf16*)(ws + 768 * 1024);             // 512*512*2 = 512 KB

    transpose_to_bf16<<<(SS * SS + 255) / 256, 256, 0, stream>>>(Wq, WqT, SS, SS);
    transpose_to_bf16<<<(SF * SS + 255) / 256, 256, 0, stream>>>(Wk, WkT, SF, SS);
    transpose_to_bf16<<<(SS * SS + 255) / 256, 256, 0, stream>>>(Wv, WvT, SS, SS);

    attn_fused<<<SB * (SS / MT), 256, 0, stream>>>(x1, x2, WqT, WkT, WvT, out);
    l2norm_inplace<<<dim3(SB, 2), 256, 0, stream>>>(out);
}
